// AdaAttModel_67053029425582
// MI455X (gfx1250) — matrix operand order repaired and checked by reading, not compiled, not run on hardware
//
#include <hip/hip_runtime.h>
#include <hip/hip_bf16.h>
#include <math.h>

// ---------------------------------------------------------------------------
// CDNA5 WMMA types (wave32): v_wmma_f32_16x16x32_bf16, f32 accumulate
// ---------------------------------------------------------------------------
typedef __attribute__((ext_vector_type(16))) __bf16 bf16x16;
typedef __attribute__((ext_vector_type(8)))  float  floatx8;

union Frag16 {
  bf16x16 v;
  __bf16  h[16];
};

__device__ __forceinline__ void cvt_frag8(Frag16& f, int base, float4 a, float4 b) {
  f.h[base + 0] = (__bf16)a.x; f.h[base + 1] = (__bf16)a.y;
  f.h[base + 2] = (__bf16)a.z; f.h[base + 3] = (__bf16)a.w;
  f.h[base + 4] = (__bf16)b.x; f.h[base + 5] = (__bf16)b.y;
  f.h[base + 6] = (__bf16)b.z; f.h[base + 7] = (__bf16)b.w;
}

// Accumulate 4 adjacent 16x16 tiles (16 rows x 64 cols) over K (mult of 32).
// A fragment (ISA 7.12.2, 16-bit A 16x32): lane m = l&15; lanes 0-15 cover
//   K = 0..15, lanes 16-31 cover K = 16..31 (contiguous order, same as B).
// B fragment (32x16): lane n = l&15; lanes 0-15 hold K=0..15, lanes 16-31
//   hold K=16..31, two packed elements per VGPR.
// All 4 B fragments are materialized in distinct registers BEFORE the 4
// back-to-back WMMAs so the next iteration's v_cvt_pk_bf16_f32 never
// immediately overwrites a fragment a just-issued WMMA read (avoids the
// 4-coexec-NOP WMMA->VALU WAR hazard of ISA 7.12.1).
__device__ __forceinline__ void gemm_phase(floatx8 acc[4],
    const float* __restrict__ Arow, const float* __restrict__ const* Brow,
    int K, int akb, int bkb) {
  for (int k0 = 0; k0 < K; k0 += 32) {
    Frag16 fa;
    {
      const float4* pa0 = (const float4*)(Arow + k0 + akb);
      const float4* pa1 = (const float4*)(Arow + k0 + 8 + akb);
      cvt_frag8(fa, 0, pa0[0], pa0[1]);
      cvt_frag8(fa, 8, pa1[0], pa1[1]);
    }
    __builtin_prefetch(Arow + k0 + 64, 0, 1);          // global_prefetch_b8
    __builtin_prefetch(Brow[0] + k0 + 64, 0, 1);
    Frag16 fb[4];
#pragma unroll
    for (int j = 0; j < 4; ++j) {
      const float4* pb = (const float4*)(Brow[j] + k0 + bkb);
      cvt_frag8(fb[j], 0, pb[0], pb[1]);
      cvt_frag8(fb[j], 8, pb[2], pb[3]);
    }
#pragma unroll
    for (int j = 0; j < 4; ++j) {
      acc[j] = __builtin_amdgcn_wmma_f32_16x16x32_bf16(
          /*neg_a=*/false, fa.v, /*neg_b=*/false, fb[j].v,
          /*c_mod=*/(short)0, acc[j], /*reuse_a=*/false, /*reuse_b=*/false);
    }
  }
}

// C[M,N] = act( A[M,K] * Bt[N,K]^T (+ A2[M,K2] * B2t[N,K2]^T) + bias[N] )
// Requirements: M % 16 == 0, K % 32 == 0, K2 % 32 == 0 (call sites comply).
// 4 waves / block; each wave computes a 16x64 output strip (4 accumulators).
template<int ACT>  // 0 = none, 1 = relu, 2 = tanh
__global__ __launch_bounds__(128)
void wmma_gemm_kernel(const float* __restrict__ A, int lda, int K,
                      const float* __restrict__ Bt, int ldb,
                      const float* __restrict__ A2, int lda2, int K2,
                      const float* __restrict__ B2t, int ldb2,
                      const float* __restrict__ bias,
                      float* __restrict__ C, int ldc, int N) {
  const int lane  = threadIdx.x & 31;
  const int wavei = threadIdx.x >> 5;
  const int nt0   = (blockIdx.x * 4 + wavei) * 4;   // first of 4 N-tiles
  const int row0  = blockIdx.y * 16;
  const int mn    = lane & 15;
  const int akb   = (lane >> 4) * 16;
  const int bkb   = (lane >> 4) * 16;

  int cols[4];
#pragma unroll
  for (int j = 0; j < 4; ++j) cols[j] = (nt0 + j) * 16 + mn;

  const float* Arow = A + (size_t)(row0 + mn) * lda;
  const float* Brow[4];
#pragma unroll
  for (int j = 0; j < 4; ++j) {   // clamp (in-bounds dup read); store is guarded
    const int cc = cols[j] < N ? cols[j] : N - 1;
    Brow[j] = Bt + (size_t)cc * ldb;
  }

  floatx8 acc[4] = {};
  gemm_phase(acc, Arow, Brow, K, akb, bkb);
  if (K2 > 0) {
    const float* Arow2 = A2 + (size_t)(row0 + mn) * lda2;
#pragma unroll
    for (int j = 0; j < 4; ++j) {
      const int cc = cols[j] < N ? cols[j] : N - 1;
      Brow[j] = B2t + (size_t)cc * ldb2;
    }
    gemm_phase(acc, Arow2, Brow, K2, akb, bkb);
  }

  const int mo = (lane >> 4) * 8;   // D layout: lanes 16-31 hold rows M=8..15
#pragma unroll
  for (int j = 0; j < 4; ++j) {
    if (cols[j] < N) {
      const float bv = (bias != nullptr) ? bias[cols[j]] : 0.f;
#pragma unroll
      for (int i = 0; i < 8; ++i) {
        float x = acc[j][i] + bv;
        if (ACT == 1) x = fmaxf(x, 0.f);
        if (ACT == 2) x = tanhf(x);
        C[(size_t)(row0 + mo + i) * ldc + cols[j]] = x;
      }
    }
  }
}

// ---------------------------------------------------------------------------
// Elementwise / reduction kernels
// ---------------------------------------------------------------------------
__device__ __forceinline__ float sigf(float x) { return 1.f / (1.f + expf(-x)); }

// xt[b, 0:320] = relu(E[seq[b,t]])  (zero-padded 300 -> 320 so GEMM K%32==0)
__global__ __launch_bounds__(256)
void embed_relu_kernel(const int* __restrict__ seq, int t,
                       const float* __restrict__ E, float* __restrict__ xt) {
  const int i = blockIdx.x * 256 + threadIdx.x;
  if (i >= 128 * 320) return;
  const int b = i / 320, d = i - b * 320;
  float v = 0.f;
  if (d < 300) {
    const int tok = seq[b * 21 + t];
    const float x = E[(size_t)tok * 300 + d];
    v = x > 0.f ? x : 0.f;
  }
  xt[i] = v;
}

// zero-padded copy of w_ih: [2560,300] -> [2560,320]
__global__ __launch_bounds__(256)
void pad_wih_kernel(const float* __restrict__ w_ih, float* __restrict__ w_pad) {
  const int i = blockIdx.x * 256 + threadIdx.x;
  if (i >= 2560 * 320) return;
  const int r = i / 320, d = i - r * 320;
  w_pad[i] = (d < 300) ? w_ih[(size_t)r * 300 + d] : 0.f;
}

// gates[b, 0:5R] split i,f,g,o,s ; cy = tanh(sig(f)*cx + sig(i)*tanh(g))
__global__ __launch_bounds__(256)
void lstm_cell_kernel(const float* __restrict__ gates, const float* __restrict__ cx,
                      float* __restrict__ hy, float* __restrict__ cy,
                      float* __restrict__ sent) {
  const int i = blockIdx.x * 256 + threadIdx.x;
  if (i >= 128 * 512) return;
  const int b = i >> 9, r = i & 511;
  const float* g = gates + (size_t)b * 2560;
  const float ig = sigf(g[r]);
  const float fg = sigf(g[512 + r]);
  const float gg = tanhf(g[1024 + r]);
  const float og = sigf(g[1536 + r]);
  const float sg = sigf(g[2048 + r]);
  const float c = tanhf(fg * cx[i] + ig * gg);
  cy[i]   = c;
  sent[i] = sg * c;
  hy[i]   = og * c;
}

// e[b,a] = al_W . tanh(emb[b,a,:] + h_emb[b,:]) + al_b ; one wave per (b,a)
__global__ __launch_bounds__(256)
void energies_kernel(const float* __restrict__ v_emb, const float* __restrict__ sent_emb,
                     const float* __restrict__ h_emb, const float* __restrict__ al_W,
                     const float* __restrict__ al_b, float* __restrict__ e) {
  const int pair = blockIdx.x * 8 + (threadIdx.x >> 5);
  if (pair >= 128 * 197) return;
  const int b = pair / 197, a = pair - b * 197;
  const int lane = threadIdx.x & 31;
  const float* emb = (a == 0) ? (sent_emb + (size_t)b * 512)
                              : (v_emb + ((size_t)b * 196 + (a - 1)) * 512);
  const float* he = h_emb + (size_t)b * 512;
  float s = 0.f;
  for (int j = lane; j < 512; j += 32) s += tanhf(emb[j] + he[j]) * al_W[j];
#pragma unroll
  for (int m = 16; m > 0; m >>= 1) s += __shfl_xor(s, m, 32);
  if (lane == 0) e[pair] = s + al_b[0];
}

// softmax over 197 entries, cHat = sum alpha * [sentinel; v], x2 = cHat + hy
__global__ __launch_bounds__(256)
void softmax_ctx_kernel(const float* __restrict__ e, const float* __restrict__ v,
                        const float* __restrict__ sent, const float* __restrict__ hy,
                        float* __restrict__ x2) {
  __shared__ float sa[197];
  __shared__ float red[256];
  const int b = blockIdx.x, tid = threadIdx.x;
  float mx = -INFINITY;
  for (int a = tid; a < 197; a += 256) { float x = e[b * 197 + a]; sa[a] = x; mx = fmaxf(mx, x); }
  red[tid] = mx; __syncthreads();
  for (int s = 128; s > 0; s >>= 1) { if (tid < s) red[tid] = fmaxf(red[tid], red[tid + s]); __syncthreads(); }
  mx = red[0]; __syncthreads();
  float sum = 0.f;
  for (int a = tid; a < 197; a += 256) { float ex = expf(sa[a] - mx); sa[a] = ex; sum += ex; }
  red[tid] = sum; __syncthreads();
  for (int s = 128; s > 0; s >>= 1) { if (tid < s) red[tid] += red[tid + s]; __syncthreads(); }
  const float inv = 1.f / red[0];
  __syncthreads();
  for (int r = tid; r < 512; r += 256) {
    float acc = sa[0] * sent[b * 512 + r];
    const float* vb = v + (size_t)b * 196 * 512 + r;
#pragma unroll 4
    for (int a = 0; a < 196; ++a) acc += sa[a + 1] * vb[(size_t)a * 512];
    x2[b * 512 + r] = acc * inv + hy[b * 512 + r];
  }
}

// logp[b, t, :] = logits - logsumexp(logits) written straight into output
__global__ __launch_bounds__(256)
void logsoftmax_kernel(const float* __restrict__ logits, float* __restrict__ out, int t) {
  __shared__ float red[256];
  const int b = blockIdx.x, tid = threadIdx.x;
  const float* row = logits + (size_t)b * 7800;
  float mx = -INFINITY;
  for (int i = tid; i < 7800; i += 256) mx = fmaxf(mx, row[i]);
  red[tid] = mx; __syncthreads();
  for (int s = 128; s > 0; s >>= 1) { if (tid < s) red[tid] = fmaxf(red[tid], red[tid + s]); __syncthreads(); }
  mx = red[0]; __syncthreads();
  float sum = 0.f;
  for (int i = tid; i < 7800; i += 256) sum += expf(row[i] - mx);
  red[tid] = sum; __syncthreads();
  for (int s = 128; s > 0; s >>= 1) { if (tid < s) red[tid] += red[tid + s]; __syncthreads(); }
  const float lse = mx + logf(red[0]);
  float* orow = out + ((size_t)b * 20 + t) * 7800;
  for (int i = tid; i < 7800; i += 256) orow[i] = row[i] - lse;
}

// ---------------------------------------------------------------------------
// Host side
// ---------------------------------------------------------------------------
template<int ACT>
static void launch_gemm(hipStream_t s, int M, int N,
                        const float* A, int lda, int K, const float* Bt, int ldb,
                        const float* A2, int lda2, int K2, const float* B2t, int ldb2,
                        const float* bias, float* C, int ldc) {
  dim3 grid((unsigned)((N + 255) / 256), (unsigned)(M / 16));
  wmma_gemm_kernel<ACT><<<grid, dim3(128), 0, s>>>(A, lda, K, Bt, ldb,
                                                   A2, lda2, K2, B2t, ldb2,
                                                   bias, C, ldc, N);
}

extern "C" void kernel_launch(void* const* d_in, const int* in_sizes, int n_in,
                              void* d_out, int out_size, void* d_ws, size_t ws_size,
                              hipStream_t stream) {
  (void)in_sizes; (void)n_in; (void)out_size; (void)ws_size;
  const float* att   = (const float*)d_in[0];   // [128,196,2048]
  const int*   seq   = (const int*)  d_in[1];   // [128,21]
  const float* E     = (const float*)d_in[2];   // [7801,300]
  const float* w_ih  = (const float*)d_in[3];   // [2560,300]
  const float* w_hh  = (const float*)d_in[4];   // [2560,512]
  const float* ae_W  = (const float*)d_in[5];   // [512,2048]
  const float* ae_b  = (const float*)d_in[6];
  const float* c2a_W = (const float*)d_in[7];   // [512,512]
  const float* c2a_b = (const float*)d_in[8];
  const float* se_W  = (const float*)d_in[9];
  const float* se_b  = (const float*)d_in[10];
  const float* ho_W  = (const float*)d_in[11];
  const float* ho_b  = (const float*)d_in[12];
  const float* al_W  = (const float*)d_in[13];  // [1,512]
  const float* al_b  = (const float*)d_in[14];
  const float* a2h_W = (const float*)d_in[15];
  const float* a2h_b = (const float*)d_in[16];
  const float* lg_W  = (const float*)d_in[17];  // [7800,512]
  const float* lg_b  = (const float*)d_in[18];
  float* out = (float*)d_out;

  const int B = 128, A196 = 196, T = 20, R = 512, H = 512, V = 7800, FE = 2048;
  const int DP = 320;   // D=300 zero-padded to 320 (K % 32 == 0)

  // workspace carve-up (floats)
  float* p = (float*)d_ws;
  float* w_v       = p; p += (size_t)B * A196 * R;   // 12.8M
  float* w_vemb    = p; p += (size_t)B * A196 * H;   // 12.8M
  float* w_ihpad   = p; p += (size_t)5 * R * DP;     // padded w_ih
  float* w_xt      = p; p += (size_t)B * DP;
  float* w_gates   = p; p += (size_t)B * 5 * R;
  float* w_h0      = p; p += (size_t)B * R;
  float* w_h1      = p; p += (size_t)B * R;
  float* w_c0      = p; p += (size_t)B * R;
  float* w_c1      = p; p += (size_t)B * R;
  float* w_sent    = p; p += (size_t)B * R;
  float* w_sentemb = p; p += (size_t)B * H;
  float* w_hemb    = p; p += (size_t)B * H;
  float* w_e       = p; p += (size_t)B * 197;
  float* w_x2      = p; p += (size_t)B * R;
  float* w_hout    = p; p += (size_t)B * R;
  float* w_logits  = p; p += (size_t)B * V;

  hipMemsetAsync(w_h0, 0, (size_t)B * R * sizeof(float), stream);
  hipMemsetAsync(w_c0, 0, (size_t)B * R * sizeof(float), stream);
  pad_wih_kernel<<<(5 * R * DP + 255) / 256, 256, 0, stream>>>(w_ih, w_ihpad);

  // hoisted: v = relu(att . ae_W^T + ae_b) ; v_emb = v . c2a_W^T + c2a_b
  launch_gemm<1>(stream, B * A196, R, att, FE, FE, ae_W, FE,
                 nullptr, 0, 0, nullptr, 0, ae_b, w_v, R);
  launch_gemm<0>(stream, B * A196, H, w_v, R, R, c2a_W, R,
                 nullptr, 0, 0, nullptr, 0, c2a_b, w_vemb, H);

  float* hprev = w_h0; float* cprev = w_c0;
  float* hcur  = w_h1; float* ccur  = w_c1;
  for (int t = 0; t < T; ++t) {
    embed_relu_kernel<<<(B * DP + 255) / 256, 256, 0, stream>>>(seq, t, E, w_xt);
    // gates = xt.w_ih^T + h.w_hh^T  (fused dual-K GEMM)
    launch_gemm<0>(stream, B, 5 * R, w_xt, DP, DP, w_ihpad, DP,
                   hprev, R, R, w_hh, R, nullptr, w_gates, 5 * R);
    lstm_cell_kernel<<<(B * R + 255) / 256, 256, 0, stream>>>(w_gates, cprev,
                                                              hcur, ccur, w_sent);
    launch_gemm<0>(stream, B, H, w_sent, R, R, se_W, R,
                   nullptr, 0, 0, nullptr, 0, se_b, w_sentemb, H);
    launch_gemm<0>(stream, B, H, hcur, R, R, ho_W, R,
                   nullptr, 0, 0, nullptr, 0, ho_b, w_hemb, H);
    energies_kernel<<<(B * 197 + 7) / 8, 256, 0, stream>>>(w_vemb, w_sentemb,
                                                           w_hemb, al_W, al_b, w_e);
    softmax_ctx_kernel<<<B, 256, 0, stream>>>(w_e, w_v, w_sent, hcur, w_x2);
    launch_gemm<2>(stream, B, R, w_x2, R, R, a2h_W, R,
                   nullptr, 0, 0, nullptr, 0, a2h_b, w_hout, R);
    launch_gemm<0>(stream, B, V, w_hout, R, R, lg_W, R,
                   nullptr, 0, 0, nullptr, 0, lg_b, w_logits, V);
    logsoftmax_kernel<<<B, 256, 0, stream>>>(w_logits, out, t);
    float* th = hprev; hprev = hcur; hcur = th;
    float* tc = cprev; cprev = ccur; ccur = tc;
  }
}
